// GATLayer_63316407878127
// MI455X (gfx1250) — matrix operand order repaired and checked by reading, not compiled, not run on hardware
//
#include <hip/hip_runtime.h>

#define N_NODES 50000
#define N_EDGES 1600000
#define IN_F    256
#define OUT_ALL 256   // NHEAD * OUT_F
#define NHEAD   4
#define OUT_F   64
#define ALPHA_SLOPE 0.2f

typedef __bf16 bf16_t;
typedef bf16_t v16bf __attribute__((ext_vector_type(16)));
typedef float  v8f   __attribute__((ext_vector_type(8)));

// ---------------------------------------------------------------- utilities
__global__ void zero_f32(float* p, size_t n) {
    size_t i = (size_t)blockIdx.x * blockDim.x + threadIdx.x;
    size_t stride = (size_t)gridDim.x * blockDim.x;
    for (; i < n; i += stride) p[i] = 0.0f;
}

// W [K=256][N=256] fp32 row-major  ->  Wt [N=256][K=256] bf16 row-major
__global__ void wconv_transpose(const float* __restrict__ W, bf16_t* __restrict__ Wt) {
    int k = blockIdx.x;    // 0..255
    int n = threadIdx.x;   // 0..255
    Wt[n * IN_F + k] = (bf16_t)W[k * OUT_ALL + n];
}

// ---------------------------------------------------------------- GEMM (WMMA)
// h[50000x256] = x[50000x256] @ W[256x256], computed in bf16 via
// v_wmma_f32_16x16x32_bf16.  Block = 256 threads = 8 waves.
// wave -> (mslot 0..1, nslot 0..3): 16 rows x 64 cols per wave.
__global__ void __launch_bounds__(256)
gemm_wmma_bf16(const float* __restrict__ x, const bf16_t* __restrict__ Wt,
               float* __restrict__ h) {
    const int lane  = threadIdx.x & 31;
    const int wave  = threadIdx.x >> 5;
    const int mslot = wave >> 2;
    const int nslot = wave & 3;
    const int mrow  = (blockIdx.x * 2 + mslot) * 16;
    if (mrow >= N_NODES) return;            // wave-uniform: EXEC stays all-1s
    const int ncol  = nslot * 64;
    const int lm    = lane & 15;
    const int row   = mrow + lm;
    const int kbA   = (lane < 16) ? 0 : 16;  // A: K 0-15 vs 16-31
    const int kbB   = (lane < 16) ? 0 : 16;  // B: K 0-15 vs 16-31

    v8f c0 = {}, c1 = {}, c2 = {}, c3 = {};
    const float* xrow = x + (size_t)row * IN_F;
    const bf16_t* w0 = Wt + (size_t)(ncol      + lm) * IN_F + kbB;
    const bf16_t* w1 = Wt + (size_t)(ncol + 16 + lm) * IN_F + kbB;
    const bf16_t* w2 = Wt + (size_t)(ncol + 32 + lm) * IN_F + kbB;
    const bf16_t* w3 = Wt + (size_t)(ncol + 48 + lm) * IN_F + kbB;

#pragma unroll
    for (int k0 = 0; k0 < IN_F; k0 += 32) {
        // ---- A fragment: 16 bf16 per lane, converted from fp32 x
        float4 fa = *(const float4*)(xrow + k0 + kbA);
        float4 fb = *(const float4*)(xrow + k0 + kbA + 4);
        float4 fc = *(const float4*)(xrow + k0 + kbA + 8);
        float4 fd = *(const float4*)(xrow + k0 + kbA + 12);
        v16bf A;
        A[0]  = (bf16_t)fa.x; A[1]  = (bf16_t)fa.y; A[2]  = (bf16_t)fa.z; A[3]  = (bf16_t)fa.w;
        A[4]  = (bf16_t)fb.x; A[5]  = (bf16_t)fb.y; A[6]  = (bf16_t)fb.z; A[7]  = (bf16_t)fb.w;
        A[8]  = (bf16_t)fc.x; A[9]  = (bf16_t)fc.y; A[10] = (bf16_t)fc.z; A[11] = (bf16_t)fc.w;
        A[12] = (bf16_t)fd.x; A[13] = (bf16_t)fd.y; A[14] = (bf16_t)fd.z; A[15] = (bf16_t)fd.w;
        // ---- B fragments: contiguous 32B per lane from transposed W
        v16bf B0 = *(const v16bf*)(w0 + k0);
        v16bf B1 = *(const v16bf*)(w1 + k0);
        v16bf B2 = *(const v16bf*)(w2 + k0);
        v16bf B3 = *(const v16bf*)(w3 + k0);
        c0 = __builtin_amdgcn_wmma_f32_16x16x32_bf16(false, A, false, B0, (short)0, c0, false, false);
        c1 = __builtin_amdgcn_wmma_f32_16x16x32_bf16(false, A, false, B1, (short)0, c1, false, false);
        c2 = __builtin_amdgcn_wmma_f32_16x16x32_bf16(false, A, false, B2, (short)0, c2, false, false);
        c3 = __builtin_amdgcn_wmma_f32_16x16x32_bf16(false, A, false, B3, (short)0, c3, false, false);
    }

    // C/D layout: lane holds column (lane&15), rows vgpr + (lane<16 ? 0 : 8)
    const int rbase = mrow + ((lane < 16) ? 0 : 8);
#pragma unroll
    for (int j = 0; j < 8; ++j) {
        float* orow = h + (size_t)(rbase + j) * OUT_ALL + (ncol + lm);
        orow[0]  = c0[j];
        orow[16] = c1[j];
        orow[32] = c2[j];
        orow[48] = c3[j];
    }
}

// ---------------------------------------------------------------- attention logits
// al[n][hd] = sum_f a_l[hd][f] * h[n][hd*64+f]  (ar likewise). One wave per node.
__global__ void __launch_bounds__(256)
attn_logits(const float* __restrict__ h, const float* __restrict__ a_l,
            const float* __restrict__ a_r, float* __restrict__ al,
            float* __restrict__ ar) {
    int node = blockIdx.x * 8 + (threadIdx.x >> 5);
    if (node >= N_NODES) return;
    int lane = threadIdx.x & 31;           // lane covers features lane*8 .. +7
    const float* hp = h + (size_t)node * OUT_ALL + lane * 8;
    float4 u0 = *(const float4*)(hp);
    float4 u1 = *(const float4*)(hp + 4);
    const float* lp = a_l + lane * 8;
    const float* rp = a_r + lane * 8;
    float4 l0 = *(const float4*)(lp), l1 = *(const float4*)(lp + 4);
    float4 r0 = *(const float4*)(rp), r1 = *(const float4*)(rp + 4);
    float sl = u0.x*l0.x + u0.y*l0.y + u0.z*l0.z + u0.w*l0.w
             + u1.x*l1.x + u1.y*l1.y + u1.z*l1.z + u1.w*l1.w;
    float sr = u0.x*r0.x + u0.y*r0.y + u0.z*r0.z + u0.w*r0.w
             + u1.x*r1.x + u1.y*r1.y + u1.z*r1.z + u1.w*r1.w;
    // reduce within each 8-lane group (= one head)
    sl += __shfl_xor(sl, 1); sr += __shfl_xor(sr, 1);
    sl += __shfl_xor(sl, 2); sr += __shfl_xor(sr, 2);
    sl += __shfl_xor(sl, 4); sr += __shfl_xor(sr, 4);
    if ((lane & 7) == 0) {
        int hd = lane >> 3;
        al[node * NHEAD + hd] = sl;
        ar[node * NHEAD + hd] = sr;
    }
}

// ---- ordered-uint encoding of floats so atomic umax == float max.
__device__ __forceinline__ unsigned fenc(float f) {
    unsigned u = __float_as_uint(f);
    return (u & 0x80000000u) ? ~u : (u | 0x80000000u);
}
__device__ __forceinline__ float fdec(unsigned u) {
    return (u & 0x80000000u) ? __uint_as_float(u & 0x7FFFFFFFu) : __uint_as_float(~u);
}
__device__ __forceinline__ float lrelu(float s) {
    return s > 0.0f ? s : ALPHA_SLOPE * s;
}

__global__ void edge_max(const int* __restrict__ ei, const float* __restrict__ al,
                         const float* __restrict__ ar, unsigned* __restrict__ mcode) {
    int e = blockIdx.x * 256 + threadIdx.x;
    if (e >= N_EDGES) return;
    int r = ei[e], c = ei[N_EDGES + e];
    float4 a = *(const float4*)(al + (size_t)r * NHEAD);
    float4 b = *(const float4*)(ar + (size_t)c * NHEAD);
    unsigned* m = mcode + (size_t)r * NHEAD;
    __hip_atomic_fetch_max(m + 0, fenc(lrelu(a.x + b.x)), __ATOMIC_RELAXED, __HIP_MEMORY_SCOPE_AGENT);
    __hip_atomic_fetch_max(m + 1, fenc(lrelu(a.y + b.y)), __ATOMIC_RELAXED, __HIP_MEMORY_SCOPE_AGENT);
    __hip_atomic_fetch_max(m + 2, fenc(lrelu(a.z + b.z)), __ATOMIC_RELAXED, __HIP_MEMORY_SCOPE_AGENT);
    __hip_atomic_fetch_max(m + 3, fenc(lrelu(a.w + b.w)), __ATOMIC_RELAXED, __HIP_MEMORY_SCOPE_AGENT);
}

__global__ void edge_expsum(const int* __restrict__ ei, const float* __restrict__ al,
                            const float* __restrict__ ar, const unsigned* __restrict__ mcode,
                            float* __restrict__ denom, float* __restrict__ exbuf) {
    int e = blockIdx.x * 256 + threadIdx.x;
    if (e >= N_EDGES) return;
    int r = ei[e], c = ei[N_EDGES + e];
    float4 a = *(const float4*)(al + (size_t)r * NHEAD);
    float4 b = *(const float4*)(ar + (size_t)c * NHEAD);
    const unsigned* m = mcode + (size_t)r * NHEAD;
    float4 ex;
    ex.x = __expf(lrelu(a.x + b.x) - fdec(m[0]));
    ex.y = __expf(lrelu(a.y + b.y) - fdec(m[1]));
    ex.z = __expf(lrelu(a.z + b.z) - fdec(m[2]));
    ex.w = __expf(lrelu(a.w + b.w) - fdec(m[3]));
    float* d = denom + (size_t)r * NHEAD;
    __hip_atomic_fetch_add(d + 0, ex.x, __ATOMIC_RELAXED, __HIP_MEMORY_SCOPE_AGENT);
    __hip_atomic_fetch_add(d + 1, ex.y, __ATOMIC_RELAXED, __HIP_MEMORY_SCOPE_AGENT);
    __hip_atomic_fetch_add(d + 2, ex.z, __ATOMIC_RELAXED, __HIP_MEMORY_SCOPE_AGENT);
    __hip_atomic_fetch_add(d + 3, ex.w, __ATOMIC_RELAXED, __HIP_MEMORY_SCOPE_AGENT);
    *(float4*)(exbuf + (size_t)e * NHEAD) = ex;
}

// One wave per edge: lane covers 8 of the 256 output features; alpha per head.
// h and out both live in L2 (51 MB each), so gathers + fadd atomics stay on-chip.
__global__ void __launch_bounds__(256)
edge_aggregate(const int* __restrict__ ei, const float* __restrict__ exbuf,
               const float* __restrict__ denom, const float* __restrict__ h,
               float* __restrict__ out) {
    int e = blockIdx.x * 8 + (threadIdx.x >> 5);
    if (e >= N_EDGES) return;
    int lane = threadIdx.x & 31;
    int r = ei[e], c = ei[N_EDGES + e];
    int hd = lane >> 3;
    float alpha = exbuf[(size_t)e * NHEAD + hd] / denom[(size_t)r * NHEAD + hd];
    int f0 = lane * 8;
    const float* hp = h + (size_t)c * OUT_ALL + f0;
    float4 v0 = *(const float4*)(hp);
    float4 v1 = *(const float4*)(hp + 4);
    float* op = out + (size_t)r * OUT_ALL + f0;
    __hip_atomic_fetch_add(op + 0, alpha * v0.x, __ATOMIC_RELAXED, __HIP_MEMORY_SCOPE_AGENT);
    __hip_atomic_fetch_add(op + 1, alpha * v0.y, __ATOMIC_RELAXED, __HIP_MEMORY_SCOPE_AGENT);
    __hip_atomic_fetch_add(op + 2, alpha * v0.z, __ATOMIC_RELAXED, __HIP_MEMORY_SCOPE_AGENT);
    __hip_atomic_fetch_add(op + 3, alpha * v0.w, __ATOMIC_RELAXED, __HIP_MEMORY_SCOPE_AGENT);
    __hip_atomic_fetch_add(op + 4, alpha * v1.x, __ATOMIC_RELAXED, __HIP_MEMORY_SCOPE_AGENT);
    __hip_atomic_fetch_add(op + 5, alpha * v1.y, __ATOMIC_RELAXED, __HIP_MEMORY_SCOPE_AGENT);
    __hip_atomic_fetch_add(op + 6, alpha * v1.z, __ATOMIC_RELAXED, __HIP_MEMORY_SCOPE_AGENT);
    __hip_atomic_fetch_add(op + 7, alpha * v1.w, __ATOMIC_RELAXED, __HIP_MEMORY_SCOPE_AGENT);
}

// ---------------------------------------------------------------- launch
extern "C" void kernel_launch(void* const* d_in, const int* in_sizes, int n_in,
                              void* d_out, int out_size, void* d_ws, size_t ws_size,
                              hipStream_t stream) {
    const float* x   = (const float*)d_in[0];
    const int*   ei  = (const int*)d_in[1];
    const float* W   = (const float*)d_in[2];
    const float* a_l = (const float*)d_in[3];
    const float* a_r = (const float*)d_in[4];
    float* out = (float*)d_out;

    char* ws = (char*)d_ws;
    size_t off = 0;
    auto carve = [&](size_t bytes) -> void* {
        void* p = ws + off;
        off = (off + bytes + 255) & ~(size_t)255;
        return p;
    };
    bf16_t*   Wt    = (bf16_t*)  carve((size_t)IN_F * OUT_ALL * sizeof(bf16_t));
    float*    h     = (float*)   carve((size_t)N_NODES * OUT_ALL * sizeof(float));
    float*    al    = (float*)   carve((size_t)N_NODES * NHEAD * sizeof(float));
    float*    ar    = (float*)   carve((size_t)N_NODES * NHEAD * sizeof(float));
    unsigned* mcode = (unsigned*)carve((size_t)N_NODES * NHEAD * sizeof(unsigned));
    float*    denom = (float*)   carve((size_t)N_NODES * NHEAD * sizeof(float));
    float*    exbuf = (float*)   carve((size_t)N_EDGES * NHEAD * sizeof(float));

    // init: out = 0, mcode = 0 (== encoding of "most negative"), denom = 0
    zero_f32<<<2048, 256, 0, stream>>>(out, (size_t)N_NODES * OUT_ALL);
    zero_f32<<<256, 256, 0, stream>>>((float*)mcode, (size_t)N_NODES * NHEAD);
    zero_f32<<<256, 256, 0, stream>>>(denom, (size_t)N_NODES * NHEAD);

    wconv_transpose<<<IN_F, OUT_ALL, 0, stream>>>(W, Wt);

    // 32 rows per block, 50000/16 = 3125 row-tiles -> ceil(3125/2) blocks
    gemm_wmma_bf16<<<(3125 + 1) / 2, 256, 0, stream>>>(x, Wt, h);

    attn_logits<<<(N_NODES + 7) / 8, 256, 0, stream>>>(h, a_l, a_r, al, ar);

    edge_max<<<(N_EDGES + 255) / 256, 256, 0, stream>>>(ei, al, ar, mcode);
    edge_expsum<<<(N_EDGES + 255) / 256, 256, 0, stream>>>(ei, al, ar, mcode, denom, exbuf);
    edge_aggregate<<<(N_EDGES + 7) / 8, 256, 0, stream>>>(ei, exbuf, denom, h, out);
}
